// rnn_77558519432013
// MI455X (gfx1250) — matrix operand order repaired and checked by reading, not compiled, not run on hardware
//
#include <hip/hip_runtime.h>

#define S_LEN 512
#define BATCH 2048
#define IN_D  64
#define HID   128
#define BM    16            // batch rows per workgroup -> 128 workgroups
#define NTHR  128           // 4 waves

typedef __attribute__((ext_vector_type(16))) _Float16 v16h;
typedef __attribute__((ext_vector_type(4)))  _Float16 v4h;
typedef __attribute__((ext_vector_type(8)))  float    v8f;
typedef __attribute__((ext_vector_type(4)))  float    f4;

// branchless tanh: t = 2^(2*log2e*x); tanh = 1 - 2/(t+1)
__device__ __forceinline__ float fast_tanh(float v) {
  float t = __builtin_amdgcn_exp2f(v * 2.8853900817779268f);
  return 1.0f - 2.0f * __builtin_amdgcn_rcpf(t + 1.0f);
}

#define WMMA(A, B, C) __builtin_amdgcn_wmma_f32_16x16x32_f16( \
    false, (A), false, (B), (short)0, (C), false, false)

__global__ __launch_bounds__(NTHR) void rnn_scan_wmma(
    const float* __restrict__ x, const float* __restrict__ Wih,
    const float* __restrict__ bih, const float* __restrict__ Whh,
    const float* __restrict__ bhh, float* __restrict__ out)
{
  // A-fragment-packed staging. xfrag = 8-deep ring (prefetch pipeline),
  // hfrag = double buffer. [ring/buf][kt][lane][e]
  // A layout: lane = (row%16) + 16*((koff>>3)&1), e = (koff&7) + 8*(koff>>4)
  __shared__ __align__(32) _Float16 xfrag[8][2][32][16];  // 16 KB
  __shared__ __align__(32) _Float16 hfrag[2][4][32][16];  //  8 KB

  const int tid  = threadIdx.x;
  const int wave = tid >> 5, lane = tid & 31;
  const int l16  = lane & 15, lhi = lane >> 4;
  const int bBase = blockIdx.x * BM;

  // ---- register-resident B fragments ----
  // wave owns columns n = wave*32 .. wave*32+31 (2 N-tiles, j=0/1)
  // B layout: lane holds col n = nt*16 + l16, k = kt*32 + 8*lhi + (e&7) + 16*(e>>3)
  v16h wihf[2][2], whhf[4][2];
  float bias[2];
  #pragma unroll
  for (int j = 0; j < 2; ++j) {
    const int n = wave * 32 + j * 16 + l16;
    bias[j] = bih[n] + bhh[n];
    #pragma unroll
    for (int kt = 0; kt < 2; ++kt) {
      const f4* p = (const f4*)(Wih + n * IN_D + kt * 32 + 8 * lhi);
      #pragma unroll
      for (int q = 0; q < 4; ++q) {
        f4 t = p[q + (q & 2)];
        #pragma unroll
        for (int e = 0; e < 4; ++e) wihf[kt][j][q * 4 + e] = (_Float16)t[e];
      }
    }
    #pragma unroll
    for (int kt = 0; kt < 4; ++kt) {
      const f4* p = (const f4*)(Whh + n * HID + kt * 32 + 8 * lhi);
      #pragma unroll
      for (int q = 0; q < 4; ++q) {
        f4 t = p[q + (q & 2)];
        #pragma unroll
        for (int e = 0; e < 4; ++e) whhf[kt][j][q * 4 + e] = (_Float16)t[e];
      }
    }
  }

  // ---- x staging geometry: thread covers 8 consecutive k of one row (32 B) ----
  const int sr = tid >> 3;          // row 0..15
  const int sq = tid & 7;           // 8-k group
  const size_t srow = (size_t)(bBase + sr) * IN_D + sq * 8;

  // 4 in-flight register slots; load issued at step s carries x_{s+8},
  // converted+stored into the ring at step s+4, consumed at step s+8.
  f4 xg[4][2];
  auto ld_slot = [&](int slot, int s) {
    const float* base = x + (size_t)s * (size_t)BATCH * IN_D + srow;
    #pragma unroll
    for (int c = 0; c < 2; ++c)
      xg[slot][c] = __builtin_nontemporal_load((const f4*)(base + c * 4));
  };
  auto st_slot = [&](int slot, int buf) {
    #pragma unroll
    for (int c = 0; c < 2; ++c) {
      const int k    = sq * 8 + c * 4;
      const int kt   = k >> 5, koff = k & 31;
      const int la   = sr | (((koff >> 3) & 1) << 4);
      const int e0   = (koff & 7) + ((koff >> 4) << 3);
      v4h v;
      #pragma unroll
      for (int e = 0; e < 4; ++e) v[e] = (_Float16)xg[slot][c][e];
      *(v4h*)&xfrag[buf][kt][la][e0] = v;
    }
  };

  // h0 = 0 in buffer 0
  for (int i = tid; i < 4 * 32 * 16; i += NTHR)
    (&hfrag[0][0][0][0])[i] = (_Float16)0;

  // prologue: directly stage x_0..x_3; issue in-flight loads for x_4..x_7
  #pragma unroll
  for (int p = 0; p < 4; ++p) { ld_slot(0, p); st_slot(0, p); }
  #pragma unroll
  for (int u = 0; u < 4; ++u) ld_slot(u, 4 + u);
  __syncthreads();

  v8f acc[2];
  const int lah = ((l16 >> 3) & 1) << 4;
  const int ew0 = (l16 & 7);

  auto step = [&](int u, int S8) {
    const int s = S8 + u;                 // u is a compile-time constant

    #pragma unroll
    for (int j = 0; j < 2; ++j) {
      #pragma unroll
      for (int i = 0; i < 8; ++i) acc[j][i] = bias[j];
    }

    // 6 K-groups (2 x-kt + 4 h-kt); A shared by both N-tiles; loads one ahead.
    v16h aA, aB;
    aA = *(const v16h*)&xfrag[u][0][lane][0];
    aB = *(const v16h*)&xfrag[u][1][lane][0];
    acc[0] = WMMA(aA, wihf[0][0], acc[0]);
    acc[1] = WMMA(aA, wihf[0][1], acc[1]);
    aA = *(const v16h*)&hfrag[u & 1][0][lane][0];
    acc[0] = WMMA(aB, wihf[1][0], acc[0]);
    acc[1] = WMMA(aB, wihf[1][1], acc[1]);
    aB = *(const v16h*)&hfrag[u & 1][1][lane][0];
    acc[0] = WMMA(aA, whhf[0][0], acc[0]);
    acc[1] = WMMA(aA, whhf[0][1], acc[1]);
    aA = *(const v16h*)&hfrag[u & 1][2][lane][0];
    acc[0] = WMMA(aB, whhf[1][0], acc[0]);
    acc[1] = WMMA(aB, whhf[1][1], acc[1]);
    aB = *(const v16h*)&hfrag[u & 1][3][lane][0];
    acc[0] = WMMA(aA, whhf[2][0], acc[0]);
    acc[1] = WMMA(aA, whhf[2][1], acc[1]);
    acc[0] = WMMA(aB, whhf[3][0], acc[0]);
    acc[1] = WMMA(aB, whhf[3][1], acc[1]);

    // h_t = tanh(acc) -> hfrag[(u+1)&1] directly in A-fragment layout.
    // Wave's columns n = wave*32 + j*16 + l16 all map to kt = wave.
    #pragma unroll
    for (int j = 0; j < 2; ++j) {
      const int e = ew0 + (j << 3);
      #pragma unroll
      for (int i = 0; i < 8; ++i) {
        const int la = (i + 8 * lhi) | lah;
        hfrag[(u + 1) & 1][wave][la][e] = (_Float16)fast_tanh(acc[j][i]);
      }
    }

    // ---- split barrier: signal once h stores are flushed, do the x staging
    // pipeline work in the signal->wait shadow, then wait. Any residual
    // loadcnt stall for the 4-step-old global loads lands here, where we
    // would be idling on sibling waves anyway.
    asm volatile("s_wait_dscnt 0x0\n\ts_barrier_signal -1" ::: "memory");
    if (s + 4 < S_LEN) {
      st_slot(u & 3, (u + 4) & 7);        // x_{s+4} -> ring (read at s+4)
      if (s + 8 < S_LEN) ld_slot(u & 3, s + 8);
    }
    asm volatile("s_barrier_wait -1" ::: "memory");
  };

  for (int S8 = 0; S8 < S_LEN; S8 += 8) {
    step(0, S8); step(1, S8); step(2, S8); step(3, S8);
    step(4, S8); step(5, S8); step(6, S8); step(7, S8);
  }

  // final hidden state, f32, straight from the last accumulators
  #pragma unroll
  for (int j = 0; j < 2; ++j) {
    const int ncol = wave * 32 + j * 16 + l16;
    #pragma unroll
    for (int i = 0; i < 8; ++i) {
      const int mrow = 8 * lhi + i;
      out[(size_t)(bBase + mrow) * HID + ncol] = fast_tanh(acc[j][i]);
    }
  }
}

extern "C" void kernel_launch(void* const* d_in, const int* in_sizes, int n_in,
                              void* d_out, int out_size, void* d_ws, size_t ws_size,
                              hipStream_t stream) {
  const float* x   = (const float*)d_in[0];   // [512,2048,64]
  const float* Wih = (const float*)d_in[1];   // [128,64]
  const float* bih = (const float*)d_in[2];   // [128]
  const float* Whh = (const float*)d_in[3];   // [128,128]
  const float* bhh = (const float*)d_in[4];   // [128]
  float* out = (float*)d_out;                 // [2048,128]
  (void)in_sizes; (void)n_in; (void)d_ws; (void)ws_size; (void)out_size;

  rnn_scan_wmma<<<dim3(BATCH / BM), dim3(NTHR), 0, stream>>>(x, Wih, bih, Whh, bhh, out);
}
